// GCN_52888227283272
// MI455X (gfx1250) — matrix operand order repaired and checked by reading, not compiled, not run on hardware
//
#include <hip/hip_runtime.h>
#include <hip/hip_bf16.h>

typedef __attribute__((ext_vector_type(16))) _Float16 v16h;
typedef __attribute__((ext_vector_type(8)))  _Float16 v8h;
typedef __attribute__((ext_vector_type(8)))  float    v8f;

// ---------------------------------------------------------------------------
// Fragment loaders (CDNA5 WMMA 16x16x32 f16 layouts, ISA 7.12.2, wave32)
//
// A (16x32, 16-bit): lane 0-15 holds row M=lane: VGPR0-3 = K 0..7, VGPR4-7 =
// K 16..23; lanes 16-31 hold K 8..15 / 24..31.  From a row-major array with
// row stride `stride` (f16 elements), that is two 16-byte loads per lane.
template <typename T>
__device__ __forceinline__ v16h load_fragA(const T* base, int row0, int k0,
                                           int stride, int ln, int kh) {
    const T* p = base + (size_t)(row0 + ln) * stride + k0 + kh * 8;
    union { v16h v; v8h h[2]; } u;
    u.h[0] = *(const v8h*)(p);
    u.h[1] = *(const v8h*)(p + 16);
    return u.v;
}

// B (32x16, 16-bit): lane 0-15 = column N=lane holding K 0..7 / 16..23,
// lanes 16-31 hold K 8..15 / 24..31 (same K order as A).  From W^T stored
// row-major [N x K] that is two 16-byte loads per lane.
template <typename T>
__device__ __forceinline__ v16h load_fragB(const T* base, int n0, int k0,
                                           int stride, int ln, int kh) {
    const T* p = base + (size_t)(n0 + ln) * stride + k0 + kh * 8;
    union { v16h v; v8h h[2]; } u;
    u.h[0] = *(const v8h*)(p);
    u.h[1] = *(const v8h*)(p + 16);
    return u.v;
}

// Float atomic-max via the signed/unsigned int trick: single non-returning
// hardware atomic (global_atomic_max_i32 / global_atomic_min_u32), no CAS loop.
__device__ __forceinline__ void atomic_max_f32(float* addr, float val) {
    if (val >= 0.0f) atomicMax((int*)addr, __float_as_int(val));
    else             atomicMin((unsigned int*)addr, __float_as_uint(val));
}

// ---------------------------------------------------------------------------
// Helper kernels
__global__ void init_neg_inf_kernel(unsigned int* buf, int n) {
    int i = blockIdx.x * blockDim.x + threadIdx.x;
    if (i < n) buf[i] = 0xFF800000u;            // -inf bit pattern
}

// finalize segment_max: -inf -> 0 (zero fill for empty segments), optional ReLU
__global__ void finalize_kernel(float* buf, int n, int do_relu) {
    int i = blockIdx.x * blockDim.x + threadIdx.x;
    if (i < n) {
        float v = buf[i];
        if (__float_as_uint(v) == 0xFF800000u) v = 0.0f;
        if (do_relu) v = fmaxf(v, 0.0f);
        buf[i] = v;
    }
}

// w [K x N] f32 (row-major)  ->  wt [N x K] f16 (row-major) == W^T
__global__ void convert_wt_kernel(const float* __restrict__ w,
                                  _Float16* __restrict__ wt, int K, int N) {
    int i = blockIdx.x * blockDim.x + threadIdx.x;
    if (i < K * N) {
        int k = i / N;
        int n = i - k * N;
        wt[(size_t)n * K + k] = (_Float16)w[i];
    }
}

// ---------------------------------------------------------------------------
// EdgeConv layer: tile of 64 edges per workgroup (256 threads = 8 wave32).
//   GEMM1: [64 x 2*CIN] @ [2*CIN x COUT] (+b1, ReLU)   -> LDS f16
//   GEMM2: [64 x COUT]  @ [COUT x COUT]  (+b2)         -> atomic scatter-max
// Wave w: m-block = w&3 (16 edges), n-blocks = half of COUT/16 per (w>>2).
template <int CIN, int COUT>
__global__ __launch_bounds__(256) void edge_conv_kernel(
    const float* __restrict__ x, const int* __restrict__ ei,
    const _Float16* __restrict__ w1t, const float* __restrict__ b1,
    const _Float16* __restrict__ w2t, const float* __restrict__ b2,
    float* __restrict__ accum, int nedges)
{
    constexpr int K1 = 2 * CIN;       // GEMM1 depth
    constexpr int NB = COUT / 16;     // n-blocks
    __shared__ __align__(16) _Float16 sh_feat[64 * K1];    // [xi, xj-xi] f16
    __shared__ __align__(16) _Float16 sh_h1[64 * COUT];    // hidden f16
    __shared__ int sh_src[64];
    __shared__ int sh_dst[64];

    const int tid  = threadIdx.x;
    const int tile = blockIdx.x * 64;

    // stage edge indices
    if (tid < 64) {
        int e = tile + tid;
        int s = 0, d = 0;
        if (e < nedges) { s = ei[e]; d = ei[nedges + e]; }
        sh_src[tid] = s;
        sh_dst[tid] = d;
    }
    __syncthreads();

    // gather x[dst], x[src] (L2-resident: node table is 25.6MB << 192MB L2),
    // build concat features [x_i, x_j - x_i] as f16 in LDS (WMMA A operand).
    for (int idx = tid; idx < 64 * CIN; idx += 256) {
        int e = idx / CIN;
        int c = idx - e * CIN;
        float xd = 0.0f, xs = 0.0f;
        if (tile + e < nedges) {
            xd = x[(size_t)sh_dst[e] * CIN + c];
            xs = x[(size_t)sh_src[e] * CIN + c];
        }
        sh_feat[e * K1 + c]       = (_Float16)xd;
        sh_feat[e * K1 + CIN + c] = (_Float16)(xs - xd);
    }
    __syncthreads();

    const int lane = tid & 31;
    const int wave = tid >> 5;
    const int ln   = lane & 15;   // N index in C/D layout, row index in A load
    const int kh   = lane >> 4;
    const int mblk = wave & 3;
    const int nb0  = (wave >> 2) * (NB / 2);

    // ---- GEMM1: h1 = relu(feat @ w1 + b1) ----
    for (int nb = nb0; nb < nb0 + NB / 2; ++nb) {
        v8f acc = {};
#pragma unroll
        for (int kb = 0; kb < K1 / 32; ++kb) {
            v16h a = load_fragA(&sh_feat[0], mblk * 16, kb * 32, K1, ln, kh);
            v16h b = load_fragB(w1t,        nb * 16,   kb * 32, K1, ln, kh);
            acc = __builtin_amdgcn_wmma_f32_16x16x32_f16(
                false, a, false, b, (short)0, acc, false, false);
        }
        float bias = b1[nb * 16 + ln];
#pragma unroll
        for (int v = 0; v < 8; ++v) {
            // C/D layout: lane 0-15 -> M = v, lanes 16-31 -> M = v + 8; N = ln
            float val = fmaxf(acc[v] + bias, 0.0f);
            sh_h1[(mblk * 16 + kh * 8 + v) * COUT + nb * 16 + ln] = (_Float16)val;
        }
    }
    __syncthreads();

    // ---- GEMM2: out = h1 @ w2 + b2, then scatter-max over dst ----
    for (int nb = nb0; nb < nb0 + NB / 2; ++nb) {
        v8f acc = {};
#pragma unroll
        for (int kb = 0; kb < COUT / 32; ++kb) {
            v16h a = load_fragA(&sh_h1[0], mblk * 16, kb * 32, COUT, ln, kh);
            v16h b = load_fragB(w2t,       nb * 16,   kb * 32, COUT, ln, kh);
            acc = __builtin_amdgcn_wmma_f32_16x16x32_f16(
                false, a, false, b, (short)0, acc, false, false);
        }
        float bias = b2[nb * 16 + ln];
#pragma unroll
        for (int v = 0; v < 8; ++v) {
            int row = mblk * 16 + kh * 8 + v;        // edge within tile
            if (tile + row < nedges) {
                float val = acc[v] + bias;
                atomic_max_f32(accum + (size_t)sh_dst[row] * COUT + nb * 16 + ln,
                               val);
            }
        }
    }
}

// ---------------------------------------------------------------------------
extern "C" void kernel_launch(void* const* d_in, const int* in_sizes, int n_in,
                              void* d_out, int out_size, void* d_ws, size_t ws_size,
                              hipStream_t stream) {
    (void)in_sizes; (void)n_in; (void)out_size; (void)ws_size;
    constexpr int N = 50000;
    constexpr int E = 800000;
    const int CIN[4]  = {32, 64, 128, 128};
    const int COUT[4] = {64, 128, 128, 128};

    const float* x0 = (const float*)d_in[0];
    const int*   ei = (const int*)d_in[1];

    // workspace layout: two N x 128 f32 node buffers + f16 transposed weights
    float* buf0 = (float*)d_ws;
    float* buf1 = buf0 + (size_t)N * 128;
    _Float16* wreg = (_Float16*)(buf1 + (size_t)N * 128);

    _Float16* w1t[4];
    _Float16* w2t[4];
    size_t woff = 0;
    for (int i = 0; i < 4; ++i) {
        w1t[i] = wreg + woff; woff += (size_t)COUT[i] * 2 * CIN[i];
        w2t[i] = wreg + woff; woff += (size_t)COUT[i] * COUT[i];
    }

    // convert + transpose weights to f16 (B operand wants N-major W^T)
    for (int i = 0; i < 4; ++i) {
        const float* w1 = (const float*)d_in[2 + 4 * i + 0];
        const float* w2 = (const float*)d_in[2 + 4 * i + 2];
        int k1 = 2 * CIN[i], n1 = COUT[i];
        int t1 = k1 * n1;
        convert_wt_kernel<<<(t1 + 255) / 256, 256, 0, stream>>>(w1, w1t[i], k1, n1);
        int k2 = COUT[i], n2 = COUT[i];
        int t2 = k2 * n2;
        convert_wt_kernel<<<(t2 + 255) / 256, 256, 0, stream>>>(w2, w2t[i], k2, n2);
    }

    // layer buffers: x -> buf0 -> buf1 -> buf0 -> d_out
    const float* xin[4] = {x0, buf0, buf1, buf0};
    float* acc[4] = {buf0, buf1, buf0, (float*)d_out};

    const int edge_blocks = (E + 63) / 64;
    for (int l = 0; l < 4; ++l) {
        const float* b1 = (const float*)d_in[2 + 4 * l + 1];
        const float* b2 = (const float*)d_in[2 + 4 * l + 3];
        int tot = N * COUT[l];
        init_neg_inf_kernel<<<(tot + 255) / 256, 256, 0, stream>>>(
            (unsigned int*)acc[l], tot);
        switch (l) {
        case 0:
            edge_conv_kernel<32, 64><<<edge_blocks, 256, 0, stream>>>(
                xin[l], ei, w1t[l], b1, w2t[l], b2, acc[l], E);
            break;
        case 1:
            edge_conv_kernel<64, 128><<<edge_blocks, 256, 0, stream>>>(
                xin[l], ei, w1t[l], b1, w2t[l], b2, acc[l], E);
            break;
        default:
            edge_conv_kernel<128, 128><<<edge_blocks, 256, 0, stream>>>(
                xin[l], ei, w1t[l], b1, w2t[l], b2, acc[l], E);
            break;
        }
        finalize_kernel<<<(tot + 255) / 256, 256, 0, stream>>>(
            acc[l], tot, (l != 3) ? 1 : 0);
    }
}
